// SparseResidualBlock_25211458027791
// MI455X (gfx1250) — compile-verified, matrix operand order checked by reading, not run on hardware
//
#include <hip/hip_runtime.h>

typedef __attribute__((ext_vector_type(16))) __bf16 v16bf;
typedef __attribute__((ext_vector_type(8)))  __bf16 v8bf;
typedef __attribute__((ext_vector_type(8)))  float  v8f;
typedef __attribute__((ext_vector_type(4)))  float  v4f;

constexpr int NROWS = 500000;
constexpr int KOFF  = 9;
constexpr int CCH   = 64;
constexpr int WAVES_PER_WG = 8;                 // 256 threads = 8 wave32
constexpr int TILES32 = NROWS / 32;             // 15625 row tiles of 32
constexpr int NWG   = (TILES32 + WAVES_PER_WG - 1) / WAVES_PER_WG;  // 1954
constexpr int WELEMS = KOFF * 2 * 4 * 32 * 16;  // 36864 bf16 per conv
constexpr int WBYTES = WELEMS * 2;              // 73728 B of B-fragments
constexpr int SPART_FLOATS = 2 * 8 * 2 * 64;    // [sum/sq][wave][half][ch]
constexpr float BN_EPS = 1e-4f;

// ---------------------------------------------------------------------------
// Repack w[K][C][C] (f32) into per-lane bf16 WMMA B-fragments:
// frag f = ((k*2 + c)*4 + t), layout frag[f][lane][j] (16 bf16 per lane),
// element j of lane (n = lane&15, h = lane>>4) holds
//   kk = (j<8) ? 8h+j : 16+8h+(j-8), ch = 32c+kk, d = 16t+n
// (mirrors the 16-bit A-matrix 16x32 VGPR layout, ISA 7.12.2)
// ---------------------------------------------------------------------------
__global__ void prep_weights(const float* __restrict__ w1,
                             const float* __restrict__ w2,
                             __bf16* __restrict__ fragA,
                             __bf16* __restrict__ fragB) {
  int e = blockIdx.x * 256 + threadIdx.x;
  if (e >= 2 * WELEMS) return;
  const float* w   = (e < WELEMS) ? w1 : w2;
  __bf16*      dst = (e < WELEMS) ? fragA : fragB;
  int q = (e < WELEMS) ? e : e - WELEMS;
  int f    = q >> 9;
  int r    = q & 511;
  int lane = r >> 4;
  int j    = r & 15;
  int k    = f >> 3;
  int c    = (f >> 2) & 1;
  int t    = f & 3;
  int n    = lane & 15;
  int h    = lane >> 4;
  int kk   = (j < 8) ? (8 * h + j) : (16 + 8 * h + (j - 8));
  dst[q] = (__bf16)w[(k * CCH + 32 * c + kk) * CCH + 16 * t + n];
}

// ---------------------------------------------------------------------------
// Bulk f32 -> bf16 conversion of the feature table (8 elements / thread).
// ---------------------------------------------------------------------------
__global__ void cvt_bf16(const float* __restrict__ x, __bf16* __restrict__ o) {
  int i = (blockIdx.x * 256 + threadIdx.x) * 8;
  v4f a = *(const v4f*)(x + i);
  v4f b = *(const v4f*)(x + i + 4);
  v8bf v;
#pragma unroll
  for (int j = 0; j < 4; ++j) { v[j] = (__bf16)a[j]; v[4 + j] = (__bf16)b[j]; }
  *(v8bf*)(o + i) = v;
}

// ---------------------------------------------------------------------------
// In-place BN + ReLU on the bf16 intermediate (8 elements / thread).
// ---------------------------------------------------------------------------
__global__ void apply_bn_relu(__bf16* __restrict__ y, const float* __restrict__ ss) {
  int i  = (blockIdx.x * 256 + threadIdx.x) * 8;
  int ch = i & 63;                       // 8 consecutive channels
  v4f sc0 = *(const v4f*)(ss + ch);
  v4f sc1 = *(const v4f*)(ss + ch + 4);
  v4f sh0 = *(const v4f*)(ss + 64 + ch);
  v4f sh1 = *(const v4f*)(ss + 64 + ch + 4);
  v8bf v = *(const v8bf*)(y + i);
#pragma unroll
  for (int j = 0; j < 4; ++j) {
    v[j]     = (__bf16)fmaxf((float)v[j]     * sc0[j] + sh0[j], 0.f);
    v[4 + j] = (__bf16)fmaxf((float)v[4 + j] * sc1[j] + sh1[j], 0.f);
  }
  *(v8bf*)(y + i) = v;
}

// ---------------------------------------------------------------------------
// Gather-GEMM: one wave computes a 32-row x 64-col tile (2 A row-groups x
// 4 N-tiles = 8 WMMAs per (k_off, chunk); B fragments amortized 2x).
// Weights staged global->LDS with async-to-LDS transfers (ASYNCcnt).
// k-loop kept rolled (unroll 1) to bound the scheduler's live set:
// loop-carried state = 64 accumulator VGPRs; per-iteration peak ~130 VGPRs,
// below the spill threshold. Latency hidden by 8 waves/WG occupancy.
// ---------------------------------------------------------------------------
template <bool OUT_BF16>
__global__ void __launch_bounds__(256)
conv_kernel(const __bf16* __restrict__ feats,
            const int*    __restrict__ nbr,
            const __bf16* __restrict__ wfrag,
            void* __restrict__ outp,
            float* __restrict__ partial) {   // [NWG][2][64]
  extern __shared__ char smem[];
  float* spart = (float*)(smem + WBYTES);

  // --- async-stage 73728 B of B-fragments into LDS (18 x b128 per thread) ---
  {
    const unsigned long long gbase = (unsigned long long)wfrag;
    const int tid = threadIdx.x;
#pragma unroll
    for (int i = 0; i < 18; ++i) {
      int off = (tid + i * 256) * 16;    // byte offset in both global and LDS
      asm volatile("global_load_async_to_lds_b128 %0, %1, %2"
                   :: "v"(off), "v"(off), "s"(gbase) : "memory");
    }
    asm volatile("s_wait_asynccnt 0x0" ::: "memory");
  }
  __syncthreads();

  const int wave = threadIdx.x >> 5;
  const int lane = threadIdx.x & 31;
  const int n    = lane & 15;
  const int h    = lane >> 4;
  const int tile = blockIdx.x * WAVES_PER_WG + wave;

  float s[4]  = {0.f, 0.f, 0.f, 0.f};
  float sq[4] = {0.f, 0.f, 0.f, 0.f};

  // wave-uniform predicate -> scalar branch, keeps EXEC all-ones (WMMA req.)
  if (__builtin_amdgcn_readfirstlane(tile) < TILES32) {
    const int base = tile * 32;
    v8f acc[2][4];
    const v8f vzero = {0.f, 0.f, 0.f, 0.f, 0.f, 0.f, 0.f, 0.f};
#pragma unroll
    for (int g = 0; g < 2; ++g)
#pragma unroll
      for (int t = 0; t < 4; ++t) acc[g][t] = vzero;

    const int nbase0 = (base + n) * KOFF;        // 32-bit nbr offsets
    const int nbase1 = (base + 16 + n) * KOFF;
    const v16bf* swv = (const v16bf*)smem;

#pragma unroll 1
    for (int k = 0; k < KOFF; ++k) {
      int r0 = nbr[nbase0 + k];
      int r1 = nbr[nbase1 + k];
      __builtin_assume(r0 >= 0);
      __builtin_assume(r1 >= 0);
      const int o0 = r0 * CCH;                   // 32-bit feature offsets
      const int o1 = r1 * CCH;
#pragma unroll
      for (int c = 0; c < 2; ++c) {
        const int b0 = 32 * c + 8 * h;           // first 8-channel strip
        v16bf a0, a1;
        ((v8bf*)&a0)[0] = *(const v8bf*)(feats + o0 + b0);
        ((v8bf*)&a0)[1] = *(const v8bf*)(feats + o0 + b0 + 16);
        ((v8bf*)&a1)[0] = *(const v8bf*)(feats + o1 + b0);
        ((v8bf*)&a1)[1] = *(const v8bf*)(feats + o1 + b0 + 16);
        const int fbase = ((k * 2 + c) * 4) * 32 + lane;
        v16bf w0 = swv[fbase];
        v16bf w1 = swv[fbase + 32];
        v16bf w2 = swv[fbase + 64];
        v16bf w3 = swv[fbase + 96];
        acc[0][0] = __builtin_amdgcn_wmma_f32_16x16x32_bf16(false, a0, false, w0, (short)0, acc[0][0], false, false);
        acc[1][0] = __builtin_amdgcn_wmma_f32_16x16x32_bf16(false, a1, false, w0, (short)0, acc[1][0], false, false);
        acc[0][1] = __builtin_amdgcn_wmma_f32_16x16x32_bf16(false, a0, false, w1, (short)0, acc[0][1], false, false);
        acc[1][1] = __builtin_amdgcn_wmma_f32_16x16x32_bf16(false, a1, false, w1, (short)0, acc[1][1], false, false);
        acc[0][2] = __builtin_amdgcn_wmma_f32_16x16x32_bf16(false, a0, false, w2, (short)0, acc[0][2], false, false);
        acc[1][2] = __builtin_amdgcn_wmma_f32_16x16x32_bf16(false, a1, false, w2, (short)0, acc[1][2], false, false);
        acc[0][3] = __builtin_amdgcn_wmma_f32_16x16x32_bf16(false, a0, false, w3, (short)0, acc[0][3], false, false);
        acc[1][3] = __builtin_amdgcn_wmma_f32_16x16x32_bf16(false, a1, false, w3, (short)0, acc[1][3], false, false);
      }
    }

    // --- store tile + per-lane channel partials (32-bit store offsets) ---
#pragma unroll
    for (int g = 0; g < 2; ++g)
#pragma unroll
      for (int t = 0; t < 4; ++t)
#pragma unroll
        for (int v = 0; v < 8; ++v) {
          float val = acc[g][t][v];
          int idx = (base + 16 * g + v + 8 * h) * CCH + 16 * t + n;
          if (OUT_BF16) ((__bf16*)outp)[idx] = (__bf16)val;
          else          ((float*)outp)[idx]  = val;
          s[t]  += val;
          sq[t] += val * val;
        }
  }

  // deterministic WG-level stats via LDS staging, fixed-order reduce
#pragma unroll
  for (int t = 0; t < 4; ++t) {
    spart[wave * 128 + h * 64 + 16 * t + n]        = s[t];
    spart[1024 + wave * 128 + h * 64 + 16 * t + n] = sq[t];
  }
  __syncthreads();

  if (threadIdx.x < 128) {
    int sel = threadIdx.x >> 6;   // 0 = sum, 1 = sumsq
    int ch  = threadIdx.x & 63;
    float tot = 0.f;
#pragma unroll
    for (int w = 0; w < 8; ++w) {
      tot += spart[sel * 1024 + w * 128 + ch];
      tot += spart[sel * 1024 + w * 128 + 64 + ch];
    }
    partial[(blockIdx.x * 2 + sel) * 64 + ch] = tot;
  }
}

// ---------------------------------------------------------------------------
// Fold per-WG partials into per-channel BN scale/shift (deterministic order).
// ---------------------------------------------------------------------------
__global__ void finalize_stats(const float* __restrict__ partial,
                               const float* __restrict__ gamma,
                               const float* __restrict__ beta,
                               float* __restrict__ ss_out) {
  int ch = threadIdx.x;   // 64 threads
  float sum = 0.f, sumsq = 0.f;
  for (int w = 0; w < NWG; ++w) {
    sum   += partial[(w * 2 + 0) * 64 + ch];
    sumsq += partial[(w * 2 + 1) * 64 + ch];
  }
  float mu    = sum / (float)NROWS;
  float var   = sumsq / (float)NROWS - mu * mu;
  float scale = gamma[ch] * rsqrtf(var + BN_EPS);
  ss_out[ch]      = scale;
  ss_out[64 + ch] = beta[ch] - mu * scale;
}

// ---------------------------------------------------------------------------
// out = relu(bn2(conv2_raw) + x), in place over d_out. float4 vectorized.
// ---------------------------------------------------------------------------
__global__ void final_add(const float* __restrict__ x,
                          const float* __restrict__ ss2,
                          float* __restrict__ out) {
  int i = (blockIdx.x * 256 + threadIdx.x) * 4;
  v4f v  = *(const v4f*)(out + i);
  v4f xv = *(const v4f*)(x + i);
  int ch = i & 63;
  v4f sc = *(const v4f*)(ss2 + ch);
  v4f sh = *(const v4f*)(ss2 + 64 + ch);
  v4f r;
#pragma unroll
  for (int j = 0; j < 4; ++j)
    r[j] = fmaxf(v[j] * sc[j] + sh[j] + xv[j], 0.f);
  *(v4f*)(out + i) = r;
}

// ---------------------------------------------------------------------------
extern "C" void kernel_launch(void* const* d_in, const int* in_sizes, int n_in,
                              void* d_out, int out_size, void* d_ws, size_t ws_size,
                              hipStream_t stream) {
  const float* x      = (const float*)d_in[0];
  const int*   nbr    = (const int*)  d_in[1];
  const float* w1     = (const float*)d_in[2];
  const float* w2     = (const float*)d_in[3];
  const float* gamma1 = (const float*)d_in[4];
  const float* beta1  = (const float*)d_in[5];
  const float* gamma2 = (const float*)d_in[6];
  const float* beta2  = (const float*)d_in[7];
  float* outf = (float*)d_out;

  // workspace layout (~130.2 MB)
  char* ws = (char*)d_ws;
  __bf16* fragA = (__bf16*)(ws);                              // 73728 B
  __bf16* fragB = (__bf16*)(ws + WBYTES);                     // 73728 B
  __bf16* xbf   = (__bf16*)(ws + 2 * WBYTES);                 // 64 MB
  __bf16* y1bf  = (__bf16*)(ws + 2 * WBYTES + (size_t)NROWS * CCH * 2);  // 64 MB
  float*  p1    = (float*)(ws + 2 * WBYTES + (size_t)NROWS * CCH * 4);
  float*  p2    = p1 + (size_t)NWG * 128;
  float*  ss1   = p2 + (size_t)NWG * 128;
  float*  ss2   = ss1 + 128;

  const size_t smem = WBYTES + SPART_FLOATS * sizeof(float);  // 81920 B

  prep_weights<<<(2 * WELEMS + 255) / 256, 256, 0, stream>>>(w1, w2, fragA, fragB);
  cvt_bf16<<<NROWS * CCH / 8 / 256, 256, 0, stream>>>(x, xbf);
  conv_kernel<true><<<NWG, 256, smem, stream>>>(xbf, nbr, fragA, (void*)y1bf, p1);
  finalize_stats<<<1, 64, 0, stream>>>(p1, gamma1, beta1, ss1);
  apply_bn_relu<<<NROWS * CCH / 8 / 256, 256, 0, stream>>>(y1bf, ss1);
  conv_kernel<false><<<NWG, 256, smem, stream>>>(y1bf, nbr, fragB, (void*)outf, p2);
  finalize_stats<<<1, 64, 0, stream>>>(p2, gamma2, beta2, ss2);
  final_add<<<NROWS * CCH / 4 / 256, 256, 0, stream>>>(x, ss2, outf);
}
